// GAT_Decoder_19911468384599
// MI455X (gfx1250) — compile-verified, matrix operand order checked by reading, not run on hardware
//
#include <hip/hip_runtime.h>
#include <hip/hip_bf16.h>
#include <cstdint>

// ---------------- problem constants (from reference setup) ----------------
#define B_TOT   256
#define NNODE   51
#define NPAD    64
#define DIM     128
#define NHEAD   8
#define HDIM    16
#define ROWS_PER_WG 16     // batch rows per workgroup in the persistent decoder

typedef __attribute__((ext_vector_type(16))) __bf16 v16bf;
typedef __attribute__((ext_vector_type(8)))  float  v8f;
typedef int v4i_vs __attribute__((vector_size(16)));   // matches builtin param type

#define AS1 __attribute__((address_space(1)))
#define AS3 __attribute__((address_space(3)))

union FragBF { v16bf v; unsigned int u[8]; uint4 q[2]; };

// ---- optional CDNA5 async global->LDS path (guarded; falls back cleanly) ----
#if defined(__has_builtin)
#if __has_builtin(__builtin_amdgcn_global_load_async_to_lds_b128)
#define HAVE_ASYNC_LDS 1
#endif
#endif

__device__ __forceinline__ void wait_asynccnt0() {
#if defined(__has_builtin) && __has_builtin(__builtin_amdgcn_s_wait_asynccnt)
    __builtin_amdgcn_s_wait_asynccnt(0);
#else
    asm volatile("s_wait_asynccnt 0" ::: "memory");
#endif
}

__device__ __forceinline__ unsigned f2bf(float x) {
    unsigned u = __float_as_uint(x);
    return (u + 0x7FFFu + ((u >> 16) & 1u)) >> 16;   // RNE f32 -> bf16
}
__device__ __forceinline__ float bf2f(unsigned short h) {
    return __uint_as_float(((unsigned)h) << 16);
}

// counter-based RNG (deterministic, pure function of counters)
__device__ __forceinline__ unsigned hash_u32(unsigned x) {
    x ^= x >> 16; x *= 0x7FEB352Du;
    x ^= x >> 15; x *= 0x846CA68Bu;
    x ^= x >> 16; return x;
}
__device__ __forceinline__ float gumbel_from(unsigned h) {
    float u = ((h >> 8) + 0.5f) * (1.0f / 16777216.0f);   // (0,1)
    return -logf(-logf(u));
}

// ---- WMMA fragment helpers (bf16 16x16x32, wave32 layouts per CDNA5 ISA) ----
// A (16x32): lane r=lane&15 holds row r; half=lane>>4 selects K sub-pattern.
__device__ __forceinline__ v16bf a_frag_lds(const float* base, int stride, int ks, int lane) {
    int r = lane & 15, half = lane >> 4;
    FragBF f;
#pragma unroll
    for (int v = 0; v < 8; ++v) {
        int k0 = ((v < 4) ? 2 * v : 8 + 2 * v) + 8 * half + 32 * ks;
        unsigned lo = f2bf(base[r * stride + k0]);
        unsigned hi = f2bf(base[r * stride + k0 + 1]);
        f.u[v] = lo | (hi << 16);
    }
    return f.v;
}
// B fragments are pre-packed into per-lane contiguous order -> two b128 loads.
__device__ __forceinline__ v16bf load_b_frag(const unsigned int* pw, int nt, int ks, int lane) {
    const unsigned int* p = pw + ((((nt << 2) + ks) << 5) + lane) * 8;
    FragBF f;
    f.q[0] = *(const uint4*)p;
    f.q[1] = *(const uint4*)(p + 4);
    return f.v;
}

// 16x16x128 (MxNxK, one nt slab) GEMM tile with TWO independent accumulators so consecutive
// v_wmma ops are independent (covers the WMMA->WMMA RAW hazard with real
// work instead of compiler-inserted v_nops).
__device__ __forceinline__ v8f gemm_tile(const float* abase, int stride,
                                         const unsigned int* pw, int nt, int lane) {
    v8f acc0 = {0.f, 0.f, 0.f, 0.f, 0.f, 0.f, 0.f, 0.f};
    v8f acc1 = {0.f, 0.f, 0.f, 0.f, 0.f, 0.f, 0.f, 0.f};
    v16bf a0 = a_frag_lds(abase, stride, 0, lane);
    v16bf b0 = load_b_frag(pw, nt, 0, lane);
    v16bf a1 = a_frag_lds(abase, stride, 1, lane);
    v16bf b1 = load_b_frag(pw, nt, 1, lane);
    acc0 = __builtin_amdgcn_wmma_f32_16x16x32_bf16(false, a0, false, b0, (short)0, acc0, false, false);
    acc1 = __builtin_amdgcn_wmma_f32_16x16x32_bf16(false, a1, false, b1, (short)0, acc1, false, false);
    a0 = a_frag_lds(abase, stride, 2, lane);
    b0 = load_b_frag(pw, nt, 2, lane);
    a1 = a_frag_lds(abase, stride, 3, lane);
    b1 = load_b_frag(pw, nt, 3, lane);
    acc0 = __builtin_amdgcn_wmma_f32_16x16x32_bf16(false, a0, false, b0, (short)0, acc0, false, false);
    acc1 = __builtin_amdgcn_wmma_f32_16x16x32_bf16(false, a1, false, b1, (short)0, acc1, false, false);
    return acc0 + acc1;
}

// =======================================================================
// Kernel A: convert + pack the 7 weight matrices into WMMA B-fragment
// order (bf16). Layout per weight: u32[nt(8)][ks(4)][lane(32)][8] = 32KB.
// =======================================================================
__global__ void pack_weights(const float* __restrict__ fc, const float* __restrict__ fc1,
                             const float* __restrict__ wq, const float* __restrict__ wk,
                             const float* __restrict__ wv, const float* __restrict__ wo,
                             const float* __restrict__ wkp, unsigned int* __restrict__ packed) {
    const float* W;
    switch (blockIdx.x) {
        case 0: W = fc;  break;
        case 1: W = fc1; break;
        case 2: W = wq;  break;
        case 3: W = wk;  break;
        case 4: W = wv;  break;
        case 5: W = wo;  break;
        default: W = wkp; break;
    }
    unsigned int* out = packed + (size_t)blockIdx.x * 8192;
    int t = threadIdx.x;                 // 0..1023
    int nt = t >> 7, ks = (t >> 5) & 3, lane = t & 31;
    int c = lane & 15, half = lane >> 4;
#pragma unroll
    for (int v = 0; v < 8; ++v) {
        int k0 = ((v < 4) ? 2 * v : 8 + 2 * v) + 8 * half + 32 * ks;
        int col = 16 * nt + c;
        unsigned lo = f2bf(W[(size_t)k0 * DIM + col]);
        unsigned hi = f2bf(W[(size_t)(k0 + 1) * DIM + col]);
        out[(size_t)t * 8 + v] = lo | (hi << 16);
    }
}

// =======================================================================
// Kernel B: step-invariant projections K = enc@wk, V = enc@wv, KP = enc@wkp
// (hoisted out of the decode loop). One block per batch row; the 51x128 enc
// tile is staged into LDS ONCE (async global->LDS when available) so the
// 8 waves x 3 mats x 4 mtiles fragment builds hit LDS instead of re-reading
// global memory 12x. M padded 51->64 (zero rows).
// =======================================================================
__global__ __launch_bounds__(256)
void precompute_kvp(const float* __restrict__ enc, const unsigned int* __restrict__ packed,
                    float* __restrict__ Kb, float* __restrict__ Vb, float* __restrict__ KPb) {
    __shared__ float enc_s[NPAD][DIM];          // 32KB
    int b = blockIdx.x;
    int tid = threadIdx.x;
    int w = tid >> 5, lane = tid & 31;
    int half = lane >> 4, c = lane & 15;
    int nt = w;

    // zero the pad rows (51..63)
    for (int e = tid; e < (NPAD - NNODE) * DIM; e += 256)
        enc_s[NNODE + e / DIM][e % DIM] = 0.f;
    // stage enc[b] (51*128 f32 = 1632 x b128) into LDS
    {
        const char* gsrc = (const char*)(enc + (size_t)b * NNODE * DIM);
        char* ldst = (char*)&enc_s[0][0];
#ifdef HAVE_ASYNC_LDS
        for (int e = tid; e < (NNODE * DIM) / 4; e += 256) {
            __builtin_amdgcn_global_load_async_to_lds_b128(
                (AS1 v4i_vs*)(void*)(gsrc + (size_t)e * 16),
                (AS3 v4i_vs*)(void*)(ldst + (size_t)e * 16),
                0, 0);
        }
        wait_asynccnt0();
#else
        for (int e = tid; e < (NNODE * DIM) / 4; e += 256)
            *(uint4*)(ldst + (size_t)e * 16) = *(const uint4*)(gsrc + (size_t)e * 16);
#endif
    }
    __syncthreads();

    for (int mat = 0; mat < 3; ++mat) {
        const unsigned int* pw = packed + (size_t)(mat == 0 ? 3 : (mat == 1 ? 4 : 6)) * 8192;
        float* outp = (mat == 0 ? Kb : (mat == 1 ? Vb : KPb)) + (size_t)b * NNODE * DIM;
        for (int mt = 0; mt < 4; ++mt) {
            v8f acc = gemm_tile(&enc_s[mt * 16][0], DIM, pw, nt, lane);
#pragma unroll
            for (int j = 0; j < 8; ++j) {
                int row = mt * 16 + j + 8 * half;
                if (row < NNODE) outp[(size_t)row * DIM + nt * 16 + c] = acc[j];
            }
        }
    }
}

// =======================================================================
// Kernel C: persistent decoder. 16 blocks x 512 threads (16 wave32's).
// Each block owns 16 batch rows and runs all n_steps in-kernel.
//   GEMM phases: waves 0..7 (and 8..15 for the pool GEMM) do WMMA M=16.
//   Per-row phases: wave r handles batch row r (softmax / sampling / masks).
// =======================================================================
__global__ __launch_bounds__(512)
void decode_kernel(const float* __restrict__ enc, const float* __restrict__ pool_in,
                   const float* __restrict__ capacity, const float* __restrict__ demand,
                   const float* __restrict__ fc_w_full, const unsigned int* __restrict__ packed,
                   const float* __restrict__ Kb, const float* __restrict__ Vb,
                   const float* __restrict__ KPb,
                   const float* __restrict__ T_p, const int* __restrict__ nsteps_p,
                   const int* __restrict__ nsamp_p, const int* __restrict__ greedy_p,
                   int* __restrict__ acts_out, float* __restrict__ d_out_f) {
    __shared__ float inp_s [ROWS_PER_WG][DIM];
    __shared__ float pool_s[ROWS_PER_WG][DIM];
    __shared__ float tmp_s [ROWS_PER_WG][DIM];   // pool-new, later glimpse
    __shared__ float dec_s [ROWS_PER_WG][DIM];   // dec, later glimpse@wo
    __shared__ float q_s   [ROWS_PER_WG][DIM];
    __shared__ unsigned short att_s[ROWS_PER_WG][NHEAD][NPAD];   // bf16 attn weights
    __shared__ unsigned char mask_s [ROWS_PER_WG][NPAD];
    __shared__ unsigned char mask1_s[ROWS_PER_WG][NPAD];
    __shared__ float dyn_s [ROWS_PER_WG];
    __shared__ float logp_s[ROWS_PER_WG];
    __shared__ int   idx_s [ROWS_PER_WG];

    const int tid = threadIdx.x;
    const int w = tid >> 5, lane = tid & 31;
    const int b0 = blockIdx.x * ROWS_PER_WG;
    const float T = T_p[0];
    const int n_steps = nsteps_p[0];
    const int n_samp  = nsamp_p[0];
    const int greedy  = greedy_p[0];
    const float cap0 = capacity[0];
    const float inv_sqrt_hd = 0.25f;                 // 16^-0.5
    const float inv_sqrt_h  = 0.08838834764831845f;  // 128^-0.5

    for (int smp = 0; smp < n_samp; ++smp) {
        // -------- per-sample state init (idx0 = 0 mask update baked in) -----
        for (int e = tid; e < ROWS_PER_WG * DIM; e += blockDim.x) {
            int r = e >> 7, d = e & 127;
            inp_s[r][d]  = enc[((size_t)(b0 + r) * NNODE) * DIM + d];
            pool_s[r][d] = pool_in[(size_t)(b0 + r) * DIM + d];
        }
        for (int e = tid; e < ROWS_PER_WG * NPAD; e += blockDim.x) {
            int r = e >> 6, n = e & 63;
            unsigned char m1 = (n == 0) ? 1 : 0;
            mask1_s[r][n] = m1;
            unsigned char mm = m1;
            if (n > 0 && n < NNODE && demand[(size_t)(b0 + r) * NNODE + n] > cap0) mm = 1;
            if (n >= NNODE) mm = 1;
            mask_s[r][n] = (n == 0) ? 1 : mm;
        }
        if (tid < ROWS_PER_WG) { dyn_s[tid] = cap0; logp_s[tid] = 0.f; }
        __syncthreads();

        for (int s = 0; s < n_steps; ++s) {
            // ---- phase 1: dec_raw = [inp]@fc_w[:128]  ||  pool_new = pool@fc1 ----
            {
                int nt = w & 7;
                const float* abase = (w < 8) ? &inp_s[0][0] : &pool_s[0][0];
                const unsigned int* pw = packed + (size_t)((w < 8) ? 0 : 1) * 8192;
                v8f acc = gemm_tile(abase, DIM, pw, nt, lane);
                float* obase = (w < 8) ? &dec_s[0][0] : &tmp_s[0][0];
                int half = lane >> 4, c = lane & 15;
#pragma unroll
                for (int j = 0; j < 8; ++j)
                    obase[(j + 8 * half) * DIM + nt * 16 + c] = acc[j];
            }
            __syncthreads();
            // ---- phase 1b: dec = dec_raw + pool_new + dyn * fc_w[128,:] ----
            for (int e = tid; e < ROWS_PER_WG * DIM; e += blockDim.x) {
                int r = e >> 7, d = e & 127;
                float pn = tmp_s[r][d];
                dec_s[r][d] += pn + dyn_s[r] * fc_w_full[(size_t)DIM * DIM + d];
                pool_s[r][d] = pn;
            }
            __syncthreads();
            // ---- phase 2: q = dec @ wq (waves 0..7); waves 8..15 prefetch K/V ----
            if (w < 8) {
                int nt = w;
                v8f acc = gemm_tile(&dec_s[0][0], DIM, packed + (size_t)2 * 8192, nt, lane);
                int half = lane >> 4, c = lane & 15;
#pragma unroll
                for (int j = 0; j < 8; ++j) q_s[j + 8 * half][nt * 16 + c] = acc[j];
            } else {
                int r = w - 8;
                if (lane < NNODE) {
                    __builtin_prefetch(Kb + ((size_t)(b0 + r) * NNODE + lane) * DIM, 0, 1);
                    __builtin_prefetch(Vb + ((size_t)(b0 + r) * NNODE + lane) * DIM, 0, 1);
                }
            }
            __syncthreads();
            // ---- phase 3: multi-head attention, one wave per batch row ----
            {
                int r = w, b = b0 + r;
                int n0 = lane, n1 = lane + 32;
                float u0[NHEAD], u1[NHEAD];
#pragma unroll
                for (int h = 0; h < NHEAD; ++h) { u0[h] = -1e30f; u1[h] = -1e30f; }
                if (mask_s[r][n0] == 0) {
                    const float* kp = Kb + ((size_t)b * NNODE + n0) * DIM;
#pragma unroll
                    for (int h = 0; h < NHEAD; ++h) {
                        float acc = 0.f;
#pragma unroll
                        for (int d = 0; d < HDIM; d += 4) {
                            float4 kk = *(const float4*)(kp + h * HDIM + d);
                            acc += q_s[r][h * HDIM + d]     * kk.x + q_s[r][h * HDIM + d + 1] * kk.y
                                 + q_s[r][h * HDIM + d + 2] * kk.z + q_s[r][h * HDIM + d + 3] * kk.w;
                        }
                        u0[h] = acc * inv_sqrt_hd;
                    }
                }
                if (n1 < NNODE && mask_s[r][n1] == 0) {
                    const float* kp = Kb + ((size_t)b * NNODE + n1) * DIM;
#pragma unroll
                    for (int h = 0; h < NHEAD; ++h) {
                        float acc = 0.f;
#pragma unroll
                        for (int d = 0; d < HDIM; d += 4) {
                            float4 kk = *(const float4*)(kp + h * HDIM + d);
                            acc += q_s[r][h * HDIM + d]     * kk.x + q_s[r][h * HDIM + d + 1] * kk.y
                                 + q_s[r][h * HDIM + d + 2] * kk.z + q_s[r][h * HDIM + d + 3] * kk.w;
                        }
                        u1[h] = acc * inv_sqrt_hd;
                    }
                }
#pragma unroll
                for (int h = 0; h < NHEAD; ++h) {
                    float mx = fmaxf(u0[h], u1[h]);
                    for (int o = 16; o > 0; o >>= 1) mx = fmaxf(mx, __shfl_xor(mx, o, 32));
                    float e0 = expf(u0[h] - mx);
                    float e1 = (n1 < NNODE) ? expf(u1[h] - mx) : 0.f;
                    float sm = e0 + e1;
                    for (int o = 16; o > 0; o >>= 1) sm += __shfl_xor(sm, o, 32);
                    float inv = 1.f / sm;
                    att_s[r][h][n0] = (unsigned short)f2bf(e0 * inv);
                    if (n1 < NNODE) att_s[r][h][n1] = (unsigned short)f2bf(e1 * inv);
                }
                // glimpse: each lane owns 4 contiguous dims (single head per lane)
                {
                    int f = lane * 4, h = lane >> 2;
                    float g0 = 0.f, g1 = 0.f, g2 = 0.f, g3 = 0.f;
                    const float* vbase = Vb + (size_t)b * NNODE * DIM + f;
                    for (int n = 0; n < NNODE; ++n) {
                        float a = bf2f(att_s[r][h][n]);
                        float4 vv = *(const float4*)(vbase + (size_t)n * DIM);
                        g0 += a * vv.x; g1 += a * vv.y; g2 += a * vv.z; g3 += a * vv.w;
                    }
                    tmp_s[r][f] = g0; tmp_s[r][f + 1] = g1;
                    tmp_s[r][f + 2] = g2; tmp_s[r][f + 3] = g3;
                }
            }
            __syncthreads();
            // ---- phase 4: glimpse2 = glimpse @ wo ----
            if (w < 8) {
                int nt = w;
                v8f acc = gemm_tile(&tmp_s[0][0], DIM, packed + (size_t)5 * 8192, nt, lane);
                int half = lane >> 4, c = lane & 15;
#pragma unroll
                for (int j = 0; j < 8; ++j) dec_s[j + 8 * half][nt * 16 + c] = acc[j];
            } else {
                int r = w - 8;
                if (lane < NNODE)
                    __builtin_prefetch(KPb + ((size_t)(b0 + r) * NNODE + lane) * DIM, 0, 1);
            }
            __syncthreads();
            // ---- phase 5: pointer logits + sampling, one wave per row ----
            {
                int r = w, b = b0 + r;
                int n0 = lane, n1 = lane + 32;
                float l0 = -1e30f, l1 = -1e30f;
                if (mask_s[r][n0] == 0) {
                    const float* pp = KPb + ((size_t)b * NNODE + n0) * DIM;
                    float acc = 0.f;
#pragma unroll
                    for (int d = 0; d < DIM; d += 4) {
                        float4 kk = *(const float4*)(pp + d);
                        acc += dec_s[r][d] * kk.x + dec_s[r][d + 1] * kk.y
                             + dec_s[r][d + 2] * kk.z + dec_s[r][d + 3] * kk.w;
                    }
                    l0 = 10.f * tanhf(acc * inv_sqrt_h) / T;
                }
                if (n1 < NNODE && mask_s[r][n1] == 0) {
                    const float* pp = KPb + ((size_t)b * NNODE + n1) * DIM;
                    float acc = 0.f;
#pragma unroll
                    for (int d = 0; d < DIM; d += 4) {
                        float4 kk = *(const float4*)(pp + d);
                        acc += dec_s[r][d] * kk.x + dec_s[r][d + 1] * kk.y
                             + dec_s[r][d + 2] * kk.z + dec_s[r][d + 3] * kk.w;
                    }
                    l1 = 10.f * tanhf(acc * inv_sqrt_h) / T;
                }
                // log-sum-exp over nodes
                float mx = fmaxf(l0, l1);
                for (int o = 16; o > 0; o >>= 1) mx = fmaxf(mx, __shfl_xor(mx, o, 32));
                float se = expf(l0 - mx) + ((n1 < NNODE) ? expf(l1 - mx) : 0.f);
                for (int o = 16; o > 0; o >>= 1) se += __shfl_xor(se, o, 32);
                // sample (Gumbel-max == categorical) or greedy argmax
                float s0 = l0, s1 = l1;
                if (!greedy) {
                    unsigned seed = hash_u32(0x9E3779B9u ^ (unsigned)(smp * 0x632BE59B)
                                             ^ (unsigned)(s * 0x85EBCA77) ^ (unsigned)(b * 0xC2B2AE3D));
                    if (l0 > -1e29f) s0 = l0 + gumbel_from(hash_u32(seed ^ (unsigned)(n0 * 0x27D4EB2F)));
                    if (n1 < NNODE && l1 > -1e29f)
                        s1 = l1 + gumbel_from(hash_u32(seed ^ (unsigned)(n1 * 0x27D4EB2F)));
                }
                float bv; int bi;
                if (s0 >= s1) { bv = s0; bi = n0; } else { bv = s1; bi = n1; }
                for (int o = 16; o > 0; o >>= 1) {
                    float ov = __shfl_xor(bv, o, 32);
                    int   oi = __shfl_xor(bi, o, 32);
                    if (ov > bv || (ov == bv && oi < bi)) { bv = ov; bi = oi; }
                }
                int idx = bi;
                float lsel = (idx >= 32) ? l1 : l0;
                float chosen = __shfl(lsel, idx & 31, 32);
                float log_p = chosen - mx - logf(se);
                // dynamic state
                float nd = dyn_s[r] - demand[(size_t)b * NNODE + idx];
                if (idx == 0) nd = cap0;
                if (lane == 0) {
                    dyn_s[r] = nd;
                    mask1_s[r][idx] = 1;
                    idx_s[r] = idx;
                    logp_s[r] += log_p;          // is_done (>= N) is unreachable with N=51
                    acts_out[((size_t)smp * B_TOT + b) * n_steps + s] = idx;
                }
            }
            __syncthreads();
            // ---- phase 5b: mask update + next input gather ----
            {
                int r = w, b = b0 + r;
                float nd = dyn_s[r];
                int idx = idx_s[r];
                int n0 = lane, n1 = lane + 32;
                int c1 = (n0 >= 1) ? (int)mask1_s[r][n0] : 0;
                int c2 = (n1 < NNODE) ? (int)mask1_s[r][n1] : 0;
                int sm = c1 + c2;
                for (int o = 16; o > 0; o >>= 1) sm += __shfl_xor(sm, o, 32);
                unsigned char dep = (idx == 0 && sm < NNODE - 1) ? 1 : 0;
                {
                    unsigned char mm = (mask1_s[r][n0] ||
                                        demand[(size_t)b * NNODE + n0] > nd) ? 1 : 0;
                    mask_s[r][n0] = (n0 == 0) ? dep : mm;
                }
                {
                    unsigned char mm = 1;
                    if (n1 < NNODE)
                        mm = (mask1_s[r][n1] || demand[(size_t)b * NNODE + n1] > nd) ? 1 : 0;
                    mask_s[r][n1] = mm;
                }
                if (lane == 0) mask1_s[r][0] = dep;
                // inp = enc[b, idx, :]
                int f = lane * 4;
                float4 ev = *(const float4*)(enc + ((size_t)b * NNODE + idx) * DIM + f);
                inp_s[r][f] = ev.x; inp_s[r][f + 1] = ev.y;
                inp_s[r][f + 2] = ev.z; inp_s[r][f + 3] = ev.w;
            }
            __syncthreads();
        } // steps

        if (tid < ROWS_PER_WG) {
            float* logps = d_out_f + (size_t)n_samp * B_TOT * (size_t)n_steps + (size_t)smp * B_TOT;
            logps[b0 + tid] = logp_s[tid];
        }
        __syncthreads();
    } // samples
}

// =======================================================================
// host entry
// =======================================================================
extern "C" void kernel_launch(void* const* d_in, const int* in_sizes, int n_in,
                              void* d_out, int out_size, void* d_ws, size_t ws_size,
                              hipStream_t stream) {
    const float* enc  = (const float*)d_in[0];
    const float* pool = (const float*)d_in[1];
    const float* cap  = (const float*)d_in[2];
    const float* dem  = (const float*)d_in[3];
    const float* fcw  = (const float*)d_in[4];   // 129 x 128
    const float* fc1  = (const float*)d_in[5];
    const float* wq   = (const float*)d_in[6];
    const float* wk   = (const float*)d_in[7];
    const float* wv   = (const float*)d_in[8];
    const float* wo   = (const float*)d_in[9];
    const float* wkp  = (const float*)d_in[10];
    const float* T    = (const float*)d_in[11];
    const int* nsteps = (const int*)d_in[12];
    const int* nsamp  = (const int*)d_in[13];
    const int* greedy = (const int*)d_in[14];

    // workspace: 7 packed weights (7*32KB) + K, V, KP (each B*N*H f32) ~20.3MB
    unsigned int* packed = (unsigned int*)d_ws;
    float* Kb  = (float*)((char*)d_ws + 7 * 32768);
    float* Vb  = Kb + (size_t)B_TOT * NNODE * DIM;
    float* KPb = Vb + (size_t)B_TOT * NNODE * DIM;

    pack_weights<<<7, 1024, 0, stream>>>(fcw, fc1, wq, wk, wv, wo, wkp, packed);
    precompute_kvp<<<B_TOT, 256, 0, stream>>>(enc, packed, Kb, Vb, KPb);
    decode_kernel<<<B_TOT / ROWS_PER_WG, 512, 0, stream>>>(
        enc, pool, cap, dem, fcw, packed, Kb, Vb, KPb,
        T, nsteps, nsamp, greedy, (int*)d_out, (float*)d_out);
}
